// AttnDecoderRNN_89601607729486
// MI455X (gfx1250) — compile-verified, matrix operand order checked by reading, not run on hardware
//
#include <hip/hip_runtime.h>
#include <math.h>
#include <stdint.h>

// ---------------- problem constants ----------------
constexpr int cH  = 1024;   // hidden
constexpr int cO  = 10;     // output
constexpr int cS  = 64;     // encoder len
constexpr int cB  = 512;    // batch
constexpr int cM  = 10;     // motion feature dim
constexpr int cH3 = 3 * cH;           // 3072
constexpr int KC   = 128;             // K chunk staged into LDS per TDM transfer
constexpr int KPRE = 1152;            // 1034 padded up to multiple of KC (9*128)
constexpr int NPOST = 64;             // post_W rows padded 10 -> 64
constexpr int LDS_PANEL = 64 * KC;    // bf16 elements per B panel buffer (16 KB)

// ---------------- types ----------------
typedef __bf16 bf16;
typedef bf16 bf16x8  __attribute__((ext_vector_type(8)));
typedef bf16 bf16x16 __attribute__((ext_vector_type(16)));
typedef float f32x8  __attribute__((ext_vector_type(8)));
typedef unsigned int u32x4 __attribute__((ext_vector_type(4)));
typedef int i32x4 __attribute__((ext_vector_type(4)));
typedef int i32x8 __attribute__((ext_vector_type(8)));

// Load one 16-lane fragment slice: 8 contiguous bf16 at p and 8 at p+16
// (CDNA5 16-bit A/B 16x32 per-lane K layout).
__device__ __forceinline__ bf16x16 load_frag(const bf16* p) {
  bf16x8 lo = *reinterpret_cast<const bf16x8*>(p);
  bf16x8 hi = *reinterpret_cast<const bf16x8*>(p + 16);
  return __builtin_shufflevector(lo, hi, 0,1,2,3,4,5,6,7,8,9,10,11,12,13,14,15);
}

// Issue a TDM load of a 64-row x KC-col bf16 tile (row stride K elements)
// from global into LDS at byte offset lds_off.
// amdgpu-toolchain (clang-23) 6-arg builtin form.
__device__ __forceinline__ void tdm_load_b_panel(unsigned long long gaddr,
                                                 unsigned lds_off, int K) {
  u32x4 g0;
  g0[0] = 1u;                                   // count=1, user descriptor
  g0[1] = lds_off;                              // lds_addr (bytes)
  g0[2] = (unsigned)gaddr;                      // global_addr[31:0]
  g0[3] = (unsigned)((gaddr >> 32) & 0x01ffffffu) | (2u << 30); // [56:32] | type=2
  const unsigned td0 = (unsigned)K;             // tensor_dim0 (elements)
  const unsigned td1 = 64u;                     // tensor_dim1 (rows)
  const unsigned tile0 = (unsigned)KC;          // tile_dim0 (elements)
  const unsigned tile1 = 64u;                   // tile_dim1 (rows)
  const unsigned s0 = (unsigned)K;              // tensor_dim0_stride (elements)
  i32x8 g1;
  g1[0] = (int)(1u << 16);                                  // data_size=2B, no flags
  g1[1] = (int)((td0 & 0xffffu) << 16);                     // abar=0 | td0[15:0]
  g1[2] = (int)((td0 >> 16) | ((td1 & 0xffffu) << 16));     // td0[31:16] | td1[15:0]
  g1[3] = (int)((td1 >> 16) | (tile0 << 16));               // td1[31:16] | tile_dim0
  g1[4] = (int)tile1;                                       // tile_dim1 | tile_dim2=0
  g1[5] = (int)s0;                                          // stride0[31:0]
  g1[6] = 0;                                                // stride0[47:32] | stride1.lo
  g1[7] = 0;                                                // stride1.hi
  i32x4 z4 = {0, 0, 0, 0};
  i32x8 z8 = {0, 0, 0, 0, 0, 0, 0, 0};
  __builtin_amdgcn_tensor_load_to_lds(g0, g1, z4, z4, z8, 0);
}

// C[M,Nout]f32 = A[M,K]bf16 @ B[N,K]bf16^T + bias
// grid.x = N/64, grid.y = M/128, block = 256 (8 waves; wave w owns rows m0+16w..).
// B panel (64 rows) is TDM-DMA'd into LDS (double buffered, KC columns/chunk)
// and shared by all 8 waves.
__global__ void __launch_bounds__(256)
gemm_nt_bf16_wmma(const bf16* __restrict__ A, const bf16* __restrict__ Bm,
                  const float* __restrict__ bias, float* __restrict__ C,
                  int K, int Nout) {
  extern __shared__ bf16 smem[];   // 2 * LDS_PANEL bf16 = 32 KB
  const int tid  = threadIdx.x;
  const int lane = tid & 31;
  const int wave = tid >> 5;       // 0..7
  const int half = lane >> 4;      // 0 or 1
  const int l16  = lane & 15;
  const int n0 = blockIdx.x << 6;
  const int m0 = (blockIdx.y << 7) + (wave << 4);

  const bf16* Ap = A + (size_t)(m0 + l16) * K + (half << 3);
  const unsigned long long gpanel =
      (unsigned long long)(uintptr_t)(Bm + (size_t)n0 * K);

  const int nchunks = K / KC;
  if (wave == 0) tdm_load_b_panel(gpanel, 0u, K);

  f32x8 acc[4] = {};
  for (int c = 0; c < nchunks; ++c) {
    if (wave == 0) {
      if (c + 1 < nchunks) {
        tdm_load_b_panel(gpanel + (unsigned long long)(c + 1) * (KC * 2),
                         (unsigned)(((c + 1) & 1) * (LDS_PANEL * 2)), K);
        __builtin_amdgcn_s_wait_tensorcnt(1);   // chunk c landed
      } else {
        __builtin_amdgcn_s_wait_tensorcnt(0);
      }
    }
    __syncthreads();                            // panel c visible to all waves

    const bf16* bsm = smem + (size_t)(c & 1) * LDS_PANEL + (half << 3);
    const bf16* ap  = Ap + (size_t)c * KC;
    #pragma unroll
    for (int kk = 0; kk < KC; kk += 32) {
      bf16x16 a = load_frag(ap + kk);
      #pragma unroll
      for (int t = 0; t < 4; ++t) {
        bf16x16 b = load_frag(bsm + (size_t)(t * 16 + l16) * KC + kk);
        acc[t] = __builtin_amdgcn_wmma_f32_16x16x32_bf16(false, a, false, b,
                                                         (short)0, acc[t], false, false);
      }
    }
    __syncthreads();                            // done reading before buffer reuse
  }

  const int mr = m0 + (half << 3);              // first of 8 rows this lane holds
  #pragma unroll
  for (int t = 0; t < 4; ++t) {
    int n = n0 + t * 16 + l16;
    if (n < Nout) {
      float bv = bias ? bias[n] : 0.0f;
      #pragma unroll
      for (int v = 0; v < 8; ++v)
        C[(size_t)(mr + v) * Nout + n] = acc[t][v] + bv;
    }
  }
}

// ---------------- helper kernels ----------------
__global__ void cvt_bf16_kernel(const float* __restrict__ in, bf16* __restrict__ out, int n) {
  int i = blockIdx.x * blockDim.x + threadIdx.x;
  if (i < n) out[i] = (bf16)in[i];
}

// out[r*Kpad+k] = (r<R && k<K) ? in[r*K+k] : 0   (total = Rpad*Kpad)
__global__ void cvt_pad_bf16_kernel(const float* __restrict__ in, bf16* __restrict__ out,
                                    int R, int K, int Kpad, int total) {
  int i = blockIdx.x * blockDim.x + threadIdx.x;
  if (i >= total) return;
  int r = i / Kpad, k = i - r * Kpad;
  float v = (r < R && k < K) ? in[(size_t)r * K + k] : 0.0f;
  out[i] = (bf16)v;
}

// attn_WT[h*H+k] = attn_W[k*H+h]  (so q = h_last @ attn_W becomes NT GEMM)
__global__ void transpose_cvt_kernel(const float* __restrict__ in, bf16* __restrict__ out) {
  int i = blockIdx.x * blockDim.x + threadIdx.x;   // H*H
  int h = i >> 10, k = i & (cH - 1);
  out[i] = (bf16)in[(size_t)k * cH + h];
}

// energies[b,s] = enc[s,b,:] . q[b,:]   one wave per (s,b)
__global__ void __launch_bounds__(32)
energies_kernel(const float* __restrict__ enc, const float* __restrict__ q,
                float* __restrict__ e) {
  int s = blockIdx.x, b = blockIdx.y;
  const float* ep = enc + ((size_t)s * cB + b) * cH;
  const float* qp = q + (size_t)b * cH;
  float acc = 0.0f;
  for (int h = threadIdx.x; h < cH; h += 32) acc += ep[h] * qp[h];
  #pragma unroll
  for (int off = 16; off; off >>= 1) acc += __shfl_xor(acc, off, 32);
  if (threadIdx.x == 0) e[b * cS + s] = acc;
}

// softmax over S=64 per batch row; one wave per b, 2 elements/lane
__global__ void __launch_bounds__(32)
softmax_kernel(const float* __restrict__ e, float* __restrict__ w) {
  int b = blockIdx.x;
  float v0 = e[b * cS + threadIdx.x];
  float v1 = e[b * cS + threadIdx.x + 32];
  float m = fmaxf(v0, v1);
  #pragma unroll
  for (int off = 16; off; off >>= 1) m = fmaxf(m, __shfl_xor(m, off, 32));
  float e0 = __expf(v0 - m), e1 = __expf(v1 - m);
  float sum = e0 + e1;
  #pragma unroll
  for (int off = 16; off; off >>= 1) sum += __shfl_xor(sum, off, 32);
  float inv = 1.0f / sum;
  w[b * cS + threadIdx.x]      = e0 * inv;
  w[b * cS + threadIdx.x + 32] = e1 * inv;
}

// context[b,h] = sum_s w[b,s] * enc[s,b,h]
__global__ void context_kernel(const float* __restrict__ enc, const float* __restrict__ w,
                               float* __restrict__ ctx) {
  int i = blockIdx.x * blockDim.x + threadIdx.x;   // B*H
  int b = i >> 10, h = i & (cH - 1);
  const float* wb = w + b * cS;
  float acc = 0.0f;
  #pragma unroll 4
  for (int s = 0; s < cS; ++s) acc += wb[s] * enc[((size_t)s * cB + b) * cH + h];
  ctx[i] = acc;
}

// rnn_input (padded, bf16): [B, KPRE] = concat(motion[B,10], ctx[B,1024], zeros)
__global__ void build_rnn_input_kernel(const float* __restrict__ motion,
                                       const float* __restrict__ ctx,
                                       bf16* __restrict__ out) {
  int i = blockIdx.x * blockDim.x + threadIdx.x;   // B*KPRE
  if (i >= cB * KPRE) return;
  int b = i / KPRE, k = i - b * KPRE;
  float v = 0.0f;
  if (k < cM)            v = motion[b * cM + k];
  else if (k < cM + cH)  v = ctx[(size_t)b * cH + (k - cM)];
  out[i] = (bf16)v;
}

// PyTorch GRUCell gate math (r,z,n); writes fp32 hidden + bf16 copy for next GEMM
__global__ void gru_pointwise_kernel(const float* __restrict__ gi, const float* __restrict__ gh,
                                     const float* __restrict__ hprev,
                                     float* __restrict__ hout, bf16* __restrict__ hout_bf) {
  int i = blockIdx.x * blockDim.x + threadIdx.x;   // B*H
  int b = i >> 10, j = i & (cH - 1);
  size_t base = (size_t)b * cH3 + j;
  float ir = gi[base], iz = gi[base + cH], inn = gi[base + 2 * cH];
  float hr = gh[base], hz = gh[base + cH], hnn = gh[base + 2 * cH];
  float r = 1.0f / (1.0f + __expf(-(ir + hr)));
  float z = 1.0f / (1.0f + __expf(-(iz + hz)));
  float n = tanhf(inn + r * hnn);
  float h = (1.0f - z) * n + z * hprev[i];
  hout[i] = h;
  hout_bf[i] = (bf16)h;
}

// ---------------- host side ----------------
extern "C" void kernel_launch(void* const* d_in, const int* in_sizes, int n_in,
                              void* d_out, int out_size, void* d_ws, size_t ws_size,
                              hipStream_t stream) {
  (void)in_sizes; (void)n_in; (void)out_size; (void)ws_size;
  const float* motion = (const float*)d_in[0];
  const float* lasth  = (const float*)d_in[1];   // [2,B,H]
  const float* enc    = (const float*)d_in[2];   // [S,B,H]
  const float* attnW  = (const float*)d_in[3];
  // d_in[4] = attn_b: provably no effect (softmax shift invariance) -> skipped
  const float* preW   = (const float*)d_in[5];
  const float* preb   = (const float*)d_in[6];
  const float* Wih0   = (const float*)d_in[7];
  const float* Whh0   = (const float*)d_in[8];
  const float* bih0   = (const float*)d_in[9];
  const float* bhh0   = (const float*)d_in[10];
  const float* Wih1   = (const float*)d_in[11];
  const float* Whh1   = (const float*)d_in[12];
  const float* bih1   = (const float*)d_in[13];
  const float* bhh1   = (const float*)d_in[14];
  const float* postW  = (const float*)d_in[15];
  const float* postb  = (const float*)d_in[16];

  // output layout: output[B*O] | hidden[2*B*H] | attn_weights[B*S]
  float* out_lin = (float*)d_out;
  float* out_hid = out_lin + (size_t)cB * cO;
  float* out_att = out_hid + (size_t)2 * cB * cH;

  // workspace carve-up
  char* wp = (char*)d_ws;
  auto alloc = [&](size_t bytes) {
    void* p = wp; wp += (bytes + 255) & ~(size_t)255; return p;
  };
  float* q        = (float*)alloc(sizeof(float) * cB * cH);
  float* energ    = (float*)alloc(sizeof(float) * cB * cS);
  float* ctx      = (float*)alloc(sizeof(float) * cB * cH);
  float* rnn_out  = (float*)alloc(sizeof(float) * cB * cH);
  float* gi0      = (float*)alloc(sizeof(float) * cB * cH3);
  float* gh0      = (float*)alloc(sizeof(float) * cB * cH3);
  float* gi1      = (float*)alloc(sizeof(float) * cB * cH3);
  float* gh1      = (float*)alloc(sizeof(float) * cB * cH3);
  bf16* attnWT_bf = (bf16*)alloc(sizeof(bf16) * cH * cH);
  bf16* hlast1_bf = (bf16*)alloc(sizeof(bf16) * cB * cH);
  bf16* hprev0_bf = (bf16*)alloc(sizeof(bf16) * cB * cH);
  bf16* rnnin_bf  = (bf16*)alloc(sizeof(bf16) * cB * KPRE);
  bf16* preW_bf   = (bf16*)alloc(sizeof(bf16) * cH * KPRE);
  bf16* rnnout_bf = (bf16*)alloc(sizeof(bf16) * cB * cH);
  bf16* Wih0_bf   = (bf16*)alloc(sizeof(bf16) * cH3 * cH);
  bf16* Whh0_bf   = (bf16*)alloc(sizeof(bf16) * cH3 * cH);
  bf16* Wih1_bf   = (bf16*)alloc(sizeof(bf16) * cH3 * cH);
  bf16* Whh1_bf   = (bf16*)alloc(sizeof(bf16) * cH3 * cH);
  bf16* h0_bf     = (bf16*)alloc(sizeof(bf16) * cB * cH);
  bf16* h1_bf     = (bf16*)alloc(sizeof(bf16) * cB * cH);
  bf16* postW_bf  = (bf16*)alloc(sizeof(bf16) * NPOST * cH);

  const int TB = 256;
  auto nb = [](int n, int b) { return (n + b - 1) / b; };

  // --- conversions ---
  transpose_cvt_kernel<<<nb(cH * cH, TB), TB, 0, stream>>>(attnW, attnWT_bf);
  cvt_bf16_kernel<<<nb(cB * cH, TB), TB, 0, stream>>>(lasth + (size_t)cB * cH, hlast1_bf, cB * cH);
  cvt_bf16_kernel<<<nb(cB * cH, TB), TB, 0, stream>>>(lasth, hprev0_bf, cB * cH);
  cvt_bf16_kernel<<<nb(cH3 * cH, TB), TB, 0, stream>>>(Wih0, Wih0_bf, cH3 * cH);
  cvt_bf16_kernel<<<nb(cH3 * cH, TB), TB, 0, stream>>>(Whh0, Whh0_bf, cH3 * cH);
  cvt_bf16_kernel<<<nb(cH3 * cH, TB), TB, 0, stream>>>(Wih1, Wih1_bf, cH3 * cH);
  cvt_bf16_kernel<<<nb(cH3 * cH, TB), TB, 0, stream>>>(Whh1, Whh1_bf, cH3 * cH);
  cvt_pad_bf16_kernel<<<nb(cH * KPRE, TB), TB, 0, stream>>>(preW, preW_bf, cH, cH + cM, KPRE, cH * KPRE);
  cvt_pad_bf16_kernel<<<nb(NPOST * cH, TB), TB, 0, stream>>>(postW, postW_bf, cO, cH, cH, NPOST * cH);

  dim3 w256(256);
  const size_t smem = (size_t)2 * LDS_PANEL * sizeof(bf16);   // 32 KB
  // --- attention: q = h_last1 @ attn_W  (bias dropped: softmax shift-invariant) ---
  gemm_nt_bf16_wmma<<<dim3(cH / 64, cB / 128), w256, smem, stream>>>(hlast1_bf, attnWT_bf, nullptr, q, cH, cH);
  energies_kernel<<<dim3(cS, cB), dim3(32), 0, stream>>>(enc, q, energ);
  softmax_kernel<<<dim3(cB), dim3(32), 0, stream>>>(energ, out_att);
  context_kernel<<<nb(cB * cH, TB), TB, 0, stream>>>(enc, out_att, ctx);

  // --- pre-linear ---
  build_rnn_input_kernel<<<nb(cB * KPRE, TB), TB, 0, stream>>>(motion, ctx, rnnin_bf);
  gemm_nt_bf16_wmma<<<dim3(cH / 64, cB / 128), w256, smem, stream>>>(rnnin_bf, preW_bf, preb, rnn_out, KPRE, cH);
  cvt_bf16_kernel<<<nb(cB * cH, TB), TB, 0, stream>>>(rnn_out, rnnout_bf, cB * cH);

  // --- GRU layer 0 ---
  gemm_nt_bf16_wmma<<<dim3(cH3 / 64, cB / 128), w256, smem, stream>>>(rnnout_bf, Wih0_bf, bih0, gi0, cH, cH3);
  gemm_nt_bf16_wmma<<<dim3(cH3 / 64, cB / 128), w256, smem, stream>>>(hprev0_bf, Whh0_bf, bhh0, gh0, cH, cH3);
  gru_pointwise_kernel<<<nb(cB * cH, TB), TB, 0, stream>>>(gi0, gh0, lasth, out_hid, h0_bf);

  // --- GRU layer 1 ---
  gemm_nt_bf16_wmma<<<dim3(cH3 / 64, cB / 128), w256, smem, stream>>>(h0_bf, Wih1_bf, bih1, gi1, cH, cH3);
  gemm_nt_bf16_wmma<<<dim3(cH3 / 64, cB / 128), w256, smem, stream>>>(hlast1_bf, Whh1_bf, bhh1, gh1, cH, cH3);
  gru_pointwise_kernel<<<nb(cB * cH, TB), TB, 0, stream>>>(gi1, gh1, lasth + (size_t)cB * cH,
                                                           out_hid + (size_t)cB * cH, h1_bf);

  // --- post-linear (N padded to 64, stores guarded to Nout=10) ---
  gemm_nt_bf16_wmma<<<dim3(NPOST / 64, cB / 128), w256, smem, stream>>>(h1_bf, postW_bf, postb, out_lin, cH, cO);
}
